// MutualInformationLoss_55894704390786
// MI455X (gfx1250) — 8-bit matrix operand order repaired and checked by reading, compiled, not run on hardware
//
#include <hip/hip_runtime.h>

typedef __attribute__((ext_vector_type(16))) int   v16i;
typedef __attribute__((ext_vector_type(8)))  float v8f;

#define NB_COLS 4096

__device__ __forceinline__ unsigned key_of(float f) {
  unsigned u = __float_as_uint(f);
  return (u & 0x80000000u) ? ~u : (u | 0x80000000u);
}
__device__ __forceinline__ float float_of(unsigned k) {
  unsigned u = (k & 0x80000000u) ? (k & 0x7FFFFFFFu) : ~k;
  return __uint_as_float(u);
}

__device__ __forceinline__ unsigned match4(unsigned packed, unsigned rep) {
  unsigned xz = packed ^ rep;
  unsigned nz = xz + 0x7F7F7F7Fu;
  return (~nz & 0x80808080u) >> 7;
}

__device__ __forceinline__ unsigned dig(float v, float mn, float inv9) {
  int i = 1 + (int)((v - mn) * inv9);
  return (unsigned)(i > 10 ? 10 : i);
}

__global__ void mi_init_kernel(unsigned* wsu, float* wsf) {
  wsu[0] = 0xFFFFFFFFu;
  wsu[1] = 0u;
  wsu[2] = 0xFFFFFFFFu;
  wsu[3] = 0u;
  wsf[4] = 0.0f;
}

__global__ __launch_bounds__(256) void mi_minmax_kernel(
    const float* __restrict__ x, const float* __restrict__ y,
    unsigned* wsu, int n) {
  const float4* x4 = (const float4*)x;
  const float4* y4 = (const float4*)y;
  const int n4 = n >> 2;
  int i = blockIdx.x * 256 + threadIdx.x;
  const int stride = gridDim.x * 256;
  float xmn = __builtin_inff(), xmx = -__builtin_inff();
  float ymn = __builtin_inff(), ymx = -__builtin_inff();
  for (; i < n4; i += stride) {
    const float4 xv = x4[i];
    const float4 yv = y4[i];
    xmn = fminf(xmn, fminf(fminf(xv.x, xv.y), fminf(xv.z, xv.w)));
    xmx = fmaxf(xmx, fmaxf(fmaxf(xv.x, xv.y), fmaxf(xv.z, xv.w)));
    ymn = fminf(ymn, fminf(fminf(yv.x, yv.y), fminf(yv.z, yv.w)));
    ymx = fmaxf(ymx, fmaxf(fmaxf(yv.x, yv.y), fmaxf(yv.z, yv.w)));
  }
#pragma unroll
  for (int m = 1; m < 32; m <<= 1) {
    xmn = fminf(xmn, __shfl_xor(xmn, m, 32));
    xmx = fmaxf(xmx, __shfl_xor(xmx, m, 32));
    ymn = fminf(ymn, __shfl_xor(ymn, m, 32));
    ymx = fmaxf(ymx, __shfl_xor(ymx, m, 32));
  }
  if ((threadIdx.x & 31) == 0) {
    atomicMin(wsu + 0, key_of(xmn));
    atomicMax(wsu + 1, key_of(xmx));
    atomicMin(wsu + 2, key_of(ymn));
    atomicMax(wsu + 3, key_of(ymx));
  }
}

__global__ __launch_bounds__(256) void mi_hist_kernel(
    const float* __restrict__ x, const float* __restrict__ y,
    const unsigned* __restrict__ wsu, float* wsf) {
  __shared__ __align__(16) unsigned char sb[8][256];

  const int lane = threadIdx.x & 31;
  const int wave = threadIdx.x >> 5;
  const int row  = (blockIdx.x << 3) + wave;

  const float xmn  = float_of(wsu[0]);
  const float xinv = 9.0f / (float_of(wsu[1]) - xmn);
  const float ymn  = float_of(wsu[2]);
  const float yinv = 9.0f / (float_of(wsu[3]) - ymn);

  const float* xr = x + (size_t)row * NB_COLS;
  const float* yr = y + (size_t)row * NB_COLS;

  const int h = lane >> 4;
  const unsigned repm = 0x01010101u * (unsigned)(lane & 15);

  unsigned char* xb = &sb[wave][0];
  unsigned char* yb = &sb[wave][128];

  v8f acc = {0.f, 0.f, 0.f, 0.f, 0.f, 0.f, 0.f, 0.f};

  for (int c = 0; c < 32; ++c) {
    const int e = (c << 7) + (lane << 2);
    const float4 xv = *(const float4*)(xr + e);
    const float4 yv = *(const float4*)(yr + e);
    const unsigned pkx = dig(xv.x, xmn, xinv)        | (dig(xv.y, xmn, xinv) << 8) |
                         (dig(xv.z, xmn, xinv) << 16) | (dig(xv.w, xmn, xinv) << 24);
    const unsigned pky = dig(yv.x, ymn, yinv)        | (dig(yv.y, ymn, yinv) << 8) |
                         (dig(yv.z, ymn, yinv) << 16) | (dig(yv.w, ymn, yinv) << 24);
    ((unsigned*)xb)[lane] = pkx;
    ((unsigned*)yb)[lane] = pky;
    asm volatile("s_wait_dscnt 0" ::: "memory");

    v16i a, b;
#pragma unroll
    for (int v = 0; v < 16; ++v) {
      const unsigned pa = *(const unsigned*)(
          xb + (((v >> 2) << 5) + (h << 4) + ((v & 3) << 2)));
      const unsigned pb = *(const unsigned*)(
          yb + (((v >> 2) << 5) + (h << 4) + ((v & 3) << 2)));
      a[v] = (int)(match4(pa, repm) * 0x38u);
      b[v] = (int)(match4(pb, repm) * 0x38u);
    }
    acc = __builtin_amdgcn_wmma_f32_16x16x128_fp8_fp8(a, b, (short)0, acc, false, false);
    asm volatile("s_wait_dscnt 0" ::: "memory");
  }

  float s = 0.f;
#pragma unroll
  for (int g = 0; g < 8; ++g) s += acc[g];
  const float pyc = s + __shfl_xor(s, 16, 32);

  const float invD = 1.0f / (float)NB_COLS;
  float t = 0.0f;
#pragma unroll
  for (int g = 0; g < 8; ++g) {
    float r = acc[g];
    r += __shfl_xor(r, 1, 32);
    r += __shfl_xor(r, 2, 32);
    r += __shfl_xor(r, 4, 32);
    r += __shfl_xor(r, 8, 32);
    const float cnt = acc[g];
    if (cnt > 0.f) {
      t += (cnt * invD) * logf((cnt * (float)NB_COLS) / (r * pyc));
    }
  }
  t += __shfl_xor(t, 1, 32);
  t += __shfl_xor(t, 2, 32);
  t += __shfl_xor(t, 4, 32);
  t += __shfl_xor(t, 8, 32);
  t += __shfl_xor(t, 16, 32);
  if (lane == 0) atomicAdd(wsf + 4, t);
}

__global__ void mi_final_kernel(const float* wsf, float* out, float rows) {
  out[0] = rows / wsf[4];
}

extern "C" void kernel_launch(void* const* d_in, const int* in_sizes, int n_in,
                              void* d_out, int out_size, void* d_ws, size_t ws_size,
                              hipStream_t stream) {
  const float* x = (const float*)d_in[0];
  const float* y = (const float*)d_in[1];
  float* out = (float*)d_out;
  unsigned* wsu = (unsigned*)d_ws;
  float* wsf = (float*)d_ws;

  const int n = in_sizes[0];
  const int rows = n / NB_COLS;

  mi_init_kernel<<<1, 1, 0, stream>>>(wsu, wsf);
  mi_minmax_kernel<<<2048, 256, 0, stream>>>(x, y, wsu, n);
  mi_hist_kernel<<<rows / 8, 256, 0, stream>>>(x, y, wsu, wsf);
  mi_final_kernel<<<1, 1, 0, stream>>>(wsf, out, (float)rows);
}
